// WindowAttention_80290118632188
// MI455X (gfx1250) — matrix operand order repaired and checked by reading, not compiled, not run on hardware
//
#include <hip/hip_runtime.h>
#include <hip/hip_bf16.h>

typedef __attribute__((ext_vector_type(16))) _Float16 v16h;
typedef __attribute__((ext_vector_type(8)))  _Float16 v8h;
typedef __attribute__((ext_vector_type(8)))  float    v8f;

#define DIM   256
#define NTOK  49
#define HEADS 8
#define HD    32
#define NWIN  64
#define BWIN  4096

// LDS byte layout (dynamic shared):
//  [0, 131072)      scores f32 [8][64][64]; overlays x_s (64x256 f16), P (f16, pitch 128 halves), o_s (64x256 f16)
//  [131072, 229376) qkv_s f16 [64][768]
//  [229376, 262144) vT f16 [256][64]
//  [262144, 264192) rsum f32 [512]
#define SC_OFF   0
#define QKV_OFF  131072
#define VT_OFF   229376
#define RS_OFF   262144
#define SMEM_BYTES 264192

__device__ __forceinline__ v8f wmma_f16(v16h a, v16h b, v8f c) {
  return __builtin_amdgcn_wmma_f32_16x16x32_f16(false, a, false, b, (short)0, c, false, false);
}

// A-matrix (16x32 f16) fragment load, row-major source with given pitch (halves).
// Lane L: m = 16*mt + (L&15); kb = k0 + (L>=16 ? 8 : 0); halves i<8 -> k=kb+i, i>=8 -> k=kb+16+(i-8).
__device__ __forceinline__ v16h ldA(const _Float16* base, int pitch, int mt, int lane, int k0) {
  int m  = mt * 16 + (lane & 15);
  int kb = k0 + ((lane & 16) ? 8 : 0);
  const _Float16* p = base + m * pitch + kb;
  v8h lo = *(const v8h*)(p);
  v8h hi = *(const v8h*)(p + 16);
  v16h r;
#pragma unroll
  for (int i = 0; i < 8; ++i) { r[i] = lo[i]; r[i + 8] = hi[i]; }
  return r;
}

// B-matrix (32x16 f16) fragment load. Column n of B is contiguous in memory at `rowbase`
// (rowbase = &Bsrc[n][k0], 16B aligned). Lane L (n = L&15): kb = (L>=16 ? 8 : 0); i<8 -> k=kb+i, i>=8 -> k=kb+16+(i-8).
__device__ __forceinline__ v16h ldB(const _Float16* rowbase, int lane) {
  const _Float16* p = rowbase + ((lane & 16) ? 8 : 0);
  v8h lo = *(const v8h*)(p);
  v8h hi = *(const v8h*)(p + 16);
  v16h r;
#pragma unroll
  for (int i = 0; i < 8; ++i) { r[i] = lo[i]; r[i + 8] = hi[i]; }
  return r;
}

__global__ void swin_prep_weights(const float* __restrict__ qkv_w,
                                  const float* __restrict__ proj_w,
                                  _Float16* __restrict__ qkv_wh,
                                  _Float16* __restrict__ proj_wh) {
  int i = blockIdx.x * blockDim.x + threadIdx.x;
  if (i < 3 * DIM * DIM) qkv_wh[i] = (_Float16)qkv_w[i];
  if (i < DIM * DIM)     proj_wh[i] = (_Float16)proj_w[i];
}

__global__ void __launch_bounds__(256, 1)
swin_window_attn(const float* __restrict__ x,
                 const float* __restrict__ mask,
                 const float* __restrict__ qkv_b,
                 const float* __restrict__ bias_table,
                 const float* __restrict__ proj_b,
                 const int*   __restrict__ rel_index,
                 const _Float16* __restrict__ qkv_wh,
                 const _Float16* __restrict__ proj_wh,
                 float* __restrict__ out) {
  extern __shared__ char smem[];
  _Float16* sc_h  = (_Float16*)(smem + SC_OFF);
  float*    sc_f  = (float*)(smem + SC_OFF);
  _Float16* qkv_s = (_Float16*)(smem + QKV_OFF);
  _Float16* vT    = (_Float16*)(smem + VT_OFF);
  float*    rsum  = (float*)(smem + RS_OFF);

  const int b    = blockIdx.x;
  const int tid  = threadIdx.x;
  const int wave = tid >> 5;
  const int lane = tid & 31;
  const float scale = 0.17677669529663687f; // 32^-0.5

  // ---- Stage 1: x -> LDS f16 (64x256, rows >= 49 zeroed) ----
  const float* xb = x + (size_t)b * NTOK * DIM;
  for (int it = 0; it < 64; ++it) {
    int idx = tid + 256 * it;
    int row = idx >> 8, col = idx & 255;
    float v = (row < NTOK) ? xb[row * DIM + col] : 0.0f;
    sc_h[idx] = (_Float16)v;
  }
  __syncthreads();

  // ---- Stage 2: QKV GEMM (64x256) @ (256x768) via WMMA; write qkv_s (+ vT for V) ----
  for (int t = wave * 24; t < wave * 24 + 24; ++t) {
    int mt = t / 48, nt = t % 48;
    int col = nt * 16 + (lane & 15);
    v8f acc = {};
#pragma unroll
    for (int ks = 0; ks < 8; ++ks) {
      v16h a  = ldA(sc_h, 256, mt, lane, ks * 32);
      v16h bm = ldB(qkv_wh + col * 256 + ks * 32, lane);
      acc = wmma_f16(a, bm, acc);
    }
    float bias = qkv_b[col];
    float mul  = (col < DIM) ? scale : 1.0f;  // q gets head_dim^-0.5
    int mo = (lane & 16) ? 8 : 0;
#pragma unroll
    for (int i = 0; i < 8; ++i) {
      int m = mt * 16 + mo + i;
      float v = (acc[i] + bias) * mul;
      qkv_s[m * 768 + col] = (_Float16)v;
      if (col >= 2 * DIM) vT[(col - 2 * DIM) * 64 + m] = (_Float16)v;
    }
  }
  __syncthreads();  // x_s dead; sc region becomes scores

  // ---- Stage 3: per-head QK^T + bias + mask -> scores (f32 in LDS) ----
  const int h = wave;
  const int w = b & (NWIN - 1);
  const float* maskw = mask + (size_t)w * NTOK * NTOK;
#pragma unroll
  for (int mt = 0; mt < 4; ++mt) {
    for (int nt = 0; nt < 4; ++nt) {
      int n = nt * 16 + (lane & 15);
      v16h a  = ldA(qkv_s + h * HD, 768, mt, lane, 0);           // Q head h
      v16h bm = ldB(qkv_s + n * 768 + DIM + h * HD, lane);        // K^T: column n = K row n, contiguous
      v8f acc = {};
      acc = wmma_f16(a, bm, acc);
      int mo = (lane & 16) ? 8 : 0;
#pragma unroll
      for (int i = 0; i < 8; ++i) {
        int m = mt * 16 + mo + i;
        float v;
        if (n < NTOK && m < NTOK) {
          int ri = rel_index[m * NTOK + n];
          v = acc[i] + bias_table[ri * HEADS + h] + maskw[m * NTOK + n];
        } else {
          v = -1e30f;
        }
        sc_f[(h * 64 + m) * 64 + n] = v;
      }
    }
  }
  __syncthreads();

  // ---- Stage 4: softmax (2 rows per thread, in-place f32 -> f16 P, defer 1/sum) ----
  for (int rr = 0; rr < 2; ++rr) {
    int r = tid * 2 + rr;
    float*    srow = sc_f + r * 64;
    _Float16* prow = (_Float16*)(sc_f + r * 64);  // pitch 128 halves, same thread owns row
    float mx = -1e30f;
    for (int j = 0; j < 64; ++j) mx = fmaxf(mx, srow[j]);
    float s = 0.0f;
    for (int j = 0; j < 64; ++j) {
      float e = __expf(srow[j] - mx);  // ascending in-place rewrite is safe (2j <= 4j)
      s += e;
      prow[j] = (_Float16)e;
    }
    rsum[r] = 1.0f / s;
  }
  __syncthreads();

  // ---- Stage 5: AV = P @ V (K=64 -> 2 WMMA steps), scale rows by 1/sum ----
  v8f o_acc[4][2];
#pragma unroll
  for (int mt = 0; mt < 4; ++mt) {
    for (int nt = 0; nt < 2; ++nt) {
      v8f acc = {};
#pragma unroll
      for (int ks = 0; ks < 2; ++ks) {
        v16h a = ldA(sc_h + h * 64 * 128, 128, mt, lane, ks * 32);  // P head h, pitch 128 halves
        int d = h * HD + nt * 16 + (lane & 15);
        v16h bm = ldB(vT + d * 64 + ks * 32, lane);                  // V^T row = V column, contiguous
        acc = wmma_f16(a, bm, acc);
      }
      int mo = (lane & 16) ? 8 : 0;
#pragma unroll
      for (int i = 0; i < 8; ++i) acc[i] *= rsum[h * 64 + mt * 16 + mo + i];
      o_acc[mt][nt] = acc;
    }
  }
  __syncthreads();  // all waves done reading P before overwriting sc with O

  // ---- Stage 6: O -> LDS f16 (64x256) ----
  _Float16* o_s = sc_h;
#pragma unroll
  for (int mt = 0; mt < 4; ++mt) {
    for (int nt = 0; nt < 2; ++nt) {
      int col = h * HD + nt * 16 + (lane & 15);
      int mo = (lane & 16) ? 8 : 0;
#pragma unroll
      for (int i = 0; i < 8; ++i) {
        int m = mt * 16 + mo + i;
        o_s[m * 256 + col] = (_Float16)o_acc[mt][nt][i];
      }
    }
  }
  __syncthreads();

  // ---- Stage 7: proj GEMM (64x256) @ (256x256) + bias -> global f32 ----
  float* outb = out + (size_t)b * NTOK * DIM;
  for (int t = wave * 8; t < wave * 8 + 8; ++t) {
    int mt = t >> 4, nt = t & 15;
    int n = nt * 16 + (lane & 15);
    v8f acc = {};
#pragma unroll
    for (int ks = 0; ks < 8; ++ks) {
      v16h a  = ldA(o_s, 256, mt, lane, ks * 32);
      v16h bm = ldB(proj_wh + n * 256 + ks * 32, lane);
      acc = wmma_f16(a, bm, acc);
    }
    float pb = proj_b[n];
    int mo = (lane & 16) ? 8 : 0;
#pragma unroll
    for (int i = 0; i < 8; ++i) {
      int m = mt * 16 + mo + i;
      if (m < NTOK) outb[m * DIM + n] = acc[i] + pb;
    }
  }
}

extern "C" void kernel_launch(void* const* d_in, const int* in_sizes, int n_in,
                              void* d_out, int out_size, void* d_ws, size_t ws_size,
                              hipStream_t stream) {
  const float* x          = (const float*)d_in[0];
  const float* mask       = (const float*)d_in[1];
  const float* qkv_w      = (const float*)d_in[2];
  const float* qkv_b      = (const float*)d_in[3];
  const float* bias_table = (const float*)d_in[4];
  const float* proj_w     = (const float*)d_in[5];
  const float* proj_b     = (const float*)d_in[6];
  const int*   rel_index  = (const int*)d_in[7];
  float* out = (float*)d_out;

  _Float16* qkv_wh  = (_Float16*)d_ws;
  _Float16* proj_wh = qkv_wh + 3 * DIM * DIM;

  static bool attr_set = false;
  if (!attr_set) {
    hipFuncSetAttribute((const void*)swin_window_attn,
                        hipFuncAttributeMaxDynamicSharedMemorySize, SMEM_BYTES);
    attr_set = true;
  }

  swin_prep_weights<<<(3 * DIM * DIM + 255) / 256, 256, 0, stream>>>(
      qkv_w, proj_w, qkv_wh, proj_wh);

  swin_window_attn<<<BWIN, 256, SMEM_BYTES, stream>>>(
      x, mask, qkv_b, bias_table, proj_b, rel_index, qkv_wh, proj_wh, out);
}
